// MultiHeadAttention_55525337202908
// MI455X (gfx1250) — matrix operand order repaired and checked by reading, not compiled, not run on hardware
//
#include <hip/hip_runtime.h>
#include <hip/hip_bf16.h>
#include <stdint.h>

typedef __attribute__((ext_vector_type(16))) __bf16 v16bf;
typedef __attribute__((ext_vector_type(8)))  float  v8f;

#define EMBED   768
#define NHEAD   12
#define HDIM    64
#define BB      4
#define SEQ     2048
#define SCALING 0.125f   /* 64^-0.5 */

union FragBF {
  uint4 u4[2];
  v16bf v;
};

__device__ inline unsigned short f32_to_bf16_rne(float f) {
  unsigned int u = __float_as_uint(f);
  u += 0x7fffu + ((u >> 16) & 1u);
  return (unsigned short)(u >> 16);
}

// ---------------------------------------------------------------------------
// Elementwise f32 -> bf16 (round to nearest even)
// ---------------------------------------------------------------------------
__global__ __launch_bounds__(256) void convert_f32_bf16(
    const float* __restrict__ in, unsigned short* __restrict__ out, int n) {
  int i = blockIdx.x * 256 + threadIdx.x;
  if (i < n) out[i] = f32_to_bf16_rne(in[i]);
}

// ---------------------------------------------------------------------------
// Tiled bf16 WMMA GEMM:  C[M x N] = A[M x K] * Bw[N x K]^T + bias
//   mode 0: QKV projection -> split into q/k/v bf16 buffers [b][h][n][d],
//           Q scaled by SCALING
//   mode 1: output projection -> f32 C
// Block: 256 threads (8 waves), tile 128x128, K-step 32.
// Wave (wm,wn) in 4x2 grid owns a 32x64 sub-tile = 2x4 WMMA accumulators.
// ---------------------------------------------------------------------------
__global__ __launch_bounds__(256) void gemm_bf16_wmma(
    const unsigned short* __restrict__ A,
    const unsigned short* __restrict__ Bw,
    const float* __restrict__ bias,
    float* __restrict__ Cf,
    unsigned short* __restrict__ qb,
    unsigned short* __restrict__ kb,
    unsigned short* __restrict__ vb,
    int M, int Nn, int K, int mode)
{
  __shared__ unsigned short Asm[128 * 32];   // 8 KB
  __shared__ unsigned short Bsm[128 * 32];   // 8 KB

  const int tid  = threadIdx.x;
  const int lane = tid & 31;
  const int wave = tid >> 5;
  const int hf   = lane >> 4;   // half: 0 or 1
  const int r    = lane & 15;
  const int wm   = wave >> 1;   // 0..3
  const int wn   = wave & 1;    // 0..1
  const int mStart = blockIdx.x * 128;
  const int nStart = blockIdx.y * 128;

  const v8f vzero = {};
  v8f acc[2][4];
#pragma unroll
  for (int a = 0; a < 2; ++a)
#pragma unroll
    for (int b2 = 0; b2 < 4; ++b2) acc[a][b2] = vzero;

  for (int k0 = 0; k0 < K; k0 += 32) {
    __syncthreads();
    // Stage A and B tiles: 128x32 bf16 each = 512 chunks of 16B; 2 per thread.
#pragma unroll
    for (int c = 0; c < 2; ++c) {
      int idx  = tid + c * 256;
      int row  = idx >> 2;
      int col8 = (idx & 3) << 3;
      *(uint4*)(&Asm[row * 32 + col8]) =
          *(const uint4*)(&A[(size_t)(mStart + row) * K + k0 + col8]);
      *(uint4*)(&Bsm[row * 32 + col8]) =
          *(const uint4*)(&Bw[(size_t)(nStart + row) * K + k0 + col8]);
    }
    __syncthreads();

    FragBF af[2];
#pragma unroll
    for (int tm = 0; tm < 2; ++tm) {
      const unsigned short* base = &Asm[(wm * 32 + tm * 16 + r) * 32];
      af[tm].u4[0] = *(const uint4*)(base + 8 * hf);        // K = 8h..8h+7
      af[tm].u4[1] = *(const uint4*)(base + 16 + 8 * hf);   // K = 16+8h..23+8h
    }
    FragBF bf[4];
#pragma unroll
    for (int tn = 0; tn < 4; ++tn) {
      const unsigned short* base = &Bsm[(wn * 64 + tn * 16 + r) * 32];
      bf[tn].u4[0] = *(const uint4*)(base + 8 * hf);        // K = 8h..8h+7
      bf[tn].u4[1] = *(const uint4*)(base + 16 + 8 * hf);   // K = 16+8h..23+8h
    }
#pragma unroll
    for (int tm = 0; tm < 2; ++tm)
#pragma unroll
      for (int tn = 0; tn < 4; ++tn)
        acc[tm][tn] = __builtin_amdgcn_wmma_f32_16x16x32_bf16(
            false, af[tm].v, false, bf[tn].v, (short)0, acc[tm][tn], false, false);
  }

  // Epilogue
#pragma unroll
  for (int tm = 0; tm < 2; ++tm)
#pragma unroll
    for (int tn = 0; tn < 4; ++tn)
#pragma unroll
      for (int i = 0; i < 8; ++i) {
        int row = mStart + wm * 32 + tm * 16 + i + 8 * hf;
        int col = nStart + wn * 64 + tn * 16 + r;
        float val = acc[tm][tn][i] + bias[col];
        if (mode == 1) {
          Cf[(size_t)row * Nn + col] = val;
        } else {
          int c   = col / EMBED;          // 0=Q 1=K 2=V
          int rem = col - c * EMBED;
          int hh  = rem >> 6;
          int d   = rem & 63;
          int b   = row >> 11;            // row / 2048
          int n   = row & 2047;
          size_t o = (((size_t)(b * NHEAD + hh)) * SEQ + (size_t)n) * HDIM + d;
          if (c == 0)      qb[o] = f32_to_bf16_rne(val * SCALING);
          else if (c == 1) kb[o] = f32_to_bf16_rne(val);
          else             vb[o] = f32_to_bf16_rne(val);
        }
      }
}

// ---------------------------------------------------------------------------
// Flash attention: per (b,h), O = softmax(Q K^T) V   (the +1.0 in the
// reference is shift-invariant under softmax and is elided).
// Block: 256 threads = 8 waves; each wave owns 16 q rows (block = 128 rows).
// Loops over 64-key blocks; K staged row-major [key][d], V staged transposed
// [d][key] so both B-fragments are contiguous LDS reads.
// ---------------------------------------------------------------------------
__global__ __launch_bounds__(256) void attn_wmma(
    const unsigned short* __restrict__ q,   // [b*h][n][64] bf16 (pre-scaled)
    const unsigned short* __restrict__ k,   // [b*h][n][64] bf16
    const unsigned short* __restrict__ v,   // [b*h][n][64] bf16
    unsigned short* __restrict__ o)         // [b][n][768]  bf16
{
  __shared__ unsigned short Ksm[64 * 64];       // 8 KB
  __shared__ unsigned short Vsm[64 * 64];       // 8 KB (transposed: [d][key])
  __shared__ unsigned short Psm[8 * 16 * 64];   // 16 KB (per-wave P tiles)

  const int tid  = threadIdx.x;
  const int lane = tid & 31;
  const int wave = tid >> 5;
  const int hf   = lane >> 4;
  const int r    = lane & 15;

  const int bh = blockIdx.y;            // 0..47
  const int b  = bh / NHEAD;
  const int hh = bh - b * NHEAD;
  const int qBase = blockIdx.x * 128;

  const unsigned short* qp = q + (size_t)bh * SEQ * HDIM;
  const unsigned short* kp = k + (size_t)bh * SEQ * HDIM;
  const unsigned short* vp = v + (size_t)bh * SEQ * HDIM;

  // Q fragments for this wave's 16 rows (held for the whole kernel)
  const int qrow = qBase + wave * 16 + r;
  FragBF qf[2];
#pragma unroll
  for (int fi = 0; fi < 2; ++fi) {
    const unsigned short* base = qp + (size_t)qrow * HDIM + fi * 32;
    qf[fi].u4[0] = *(const uint4*)(base + 8 * hf);
    qf[fi].u4[1] = *(const uint4*)(base + 16 + 8 * hf);
  }

  const v8f vzero = {};
  v8f oacc[4];
#pragma unroll
  for (int tn = 0; tn < 4; ++tn) oacc[tn] = vzero;
  float mrow[8], lrow[8];
#pragma unroll
  for (int i = 0; i < 8; ++i) { mrow[i] = -1e30f; lrow[i] = 0.f; }

  for (int kb0 = 0; kb0 < SEQ; kb0 += 64) {
    __syncthreads();
    // Stage K (row-major) and V (transposed): 64x64 bf16, 2 chunks/thread.
#pragma unroll
    for (int c = 0; c < 2; ++c) {
      int idx  = tid + c * 256;
      int row  = idx >> 3;
      int col8 = (idx & 7) << 3;
      *(uint4*)(&Ksm[row * 64 + col8]) =
          *(const uint4*)(&kp[(size_t)(kb0 + row) * HDIM + col8]);
      uint4 vv = *(const uint4*)(&vp[(size_t)(kb0 + row) * HDIM + col8]);
      const unsigned short* sv = (const unsigned short*)&vv;
#pragma unroll
      for (int j = 0; j < 8; ++j)
        Vsm[(col8 + j) * 64 + row] = sv[j];
    }
    __syncthreads();

    // S = Q * K^T for this key block: 4 tiles of 16 keys, 2 d-steps each.
    v8f s[4];
#pragma unroll
    for (int kn = 0; kn < 4; ++kn) {
      s[kn] = vzero;
#pragma unroll
      for (int ds = 0; ds < 2; ++ds) {
        FragBF kf;
        const unsigned short* base = &Ksm[(kn * 16 + r) * 64 + ds * 32];
        kf.u4[0] = *(const uint4*)(base + 8 * hf);
        kf.u4[1] = *(const uint4*)(base + 16 + 8 * hf);
        s[kn] = __builtin_amdgcn_wmma_f32_16x16x32_bf16(
            false, qf[ds].v, false, kf.v, (short)0, s[kn], false, false);
      }
    }

    // Online softmax update (row reductions across the 16-lane half).
    float mnew[8], alpha[8], rs[8];
#pragma unroll
    for (int i = 0; i < 8; ++i) {
      float x = fmaxf(fmaxf(s[0][i], s[1][i]), fmaxf(s[2][i], s[3][i]));
      x = fmaxf(x, __shfl_xor(x, 1, 32));
      x = fmaxf(x, __shfl_xor(x, 2, 32));
      x = fmaxf(x, __shfl_xor(x, 4, 32));
      x = fmaxf(x, __shfl_xor(x, 8, 32));
      mnew[i]  = fmaxf(mrow[i], x);
      alpha[i] = __expf(mrow[i] - mnew[i]);
      mrow[i]  = mnew[i];
      rs[i]    = 0.f;
    }
#pragma unroll
    for (int kn = 0; kn < 4; ++kn)
#pragma unroll
      for (int i = 0; i < 8; ++i) {
        float p = __expf(s[kn][i] - mnew[i]);
        s[kn][i] = p;
        rs[i] += p;
      }
#pragma unroll
    for (int i = 0; i < 8; ++i) {
      float x = rs[i];
      x += __shfl_xor(x, 1, 32);
      x += __shfl_xor(x, 2, 32);
      x += __shfl_xor(x, 4, 32);
      x += __shfl_xor(x, 8, 32);
      lrow[i] = lrow[i] * alpha[i] + x;
    }
#pragma unroll
    for (int tn = 0; tn < 4; ++tn)
#pragma unroll
      for (int i = 0; i < 8; ++i) oacc[tn][i] *= alpha[i];

    // Redistribute P from C-layout to A-layout through per-wave LDS.
    unsigned short* pw = &Psm[wave * 16 * 64];
#pragma unroll
    for (int kn = 0; kn < 4; ++kn)
#pragma unroll
      for (int i = 0; i < 8; ++i)
        pw[(i + 8 * hf) * 64 + kn * 16 + r] = f32_to_bf16_rne(s[kn][i]);

    asm volatile("s_wait_dscnt 0" ::: "memory");

    FragBF pf[2];
#pragma unroll
    for (int fi = 0; fi < 2; ++fi) {
      const unsigned short* base = pw + r * 64 + fi * 32;
      pf[fi].u4[0] = *(const uint4*)(base + 8 * hf);
      pf[fi].u4[1] = *(const uint4*)(base + 16 + 8 * hf);
    }

    // O += P * V  (V transposed in LDS: contiguous keys per d row).
#pragma unroll
    for (int tn = 0; tn < 4; ++tn)
#pragma unroll
      for (int fi = 0; fi < 2; ++fi) {
        FragBF vf;
        const unsigned short* base = &Vsm[(tn * 16 + r) * 64 + fi * 32];
        vf.u4[0] = *(const uint4*)(base + 8 * hf);
        vf.u4[1] = *(const uint4*)(base + 16 + 8 * hf);
        oacc[tn] = __builtin_amdgcn_wmma_f32_16x16x32_bf16(
            false, pf[fi].v, false, vf.v, (short)0, oacc[tn], false, false);
      }
  }

  // Normalize and store to [b][n][h*64+d] bf16.
#pragma unroll
  for (int tn = 0; tn < 4; ++tn)
#pragma unroll
    for (int i = 0; i < 8; ++i) {
      int qr = qBase + wave * 16 + i + 8 * hf;
      float val = oacc[tn][i] / lrow[i];
      o[((size_t)(b * SEQ + qr)) * EMBED + hh * HDIM + tn * 16 + r] =
          f32_to_bf16_rne(val);
    }
}

// ---------------------------------------------------------------------------
extern "C" void kernel_launch(void* const* d_in, const int* in_sizes, int n_in,
                              void* d_out, int out_size, void* d_ws, size_t ws_size,
                              hipStream_t stream) {
  const float* x      = (const float*)d_in[0];  // [4][2048][768]
  const float* w_qkv  = (const float*)d_in[1];  // [2304][768]
  const float* b_qkv  = (const float*)d_in[2];  // [2304]
  const float* w_out  = (const float*)d_in[3];  // [768][768]
  const float* b_out  = (const float*)d_in[4];  // [768]
  float* out          = (float*)d_out;          // [4][2048][768]

  const size_t M  = (size_t)BB * SEQ;           // 8192
  const size_t nx = M * EMBED;                  // 6291456
  const size_t nq = nx;                         // per q/k/v buffer (48*2048*64)

  // Workspace layout (bf16 as ushort), total ~64.5 MB
  unsigned short* xbf   = (unsigned short*)d_ws;
  unsigned short* wqkvb = xbf + nx;
  unsigned short* woutb = wqkvb + (size_t)3 * EMBED * EMBED;
  unsigned short* qbuf  = woutb + (size_t)EMBED * EMBED;
  unsigned short* kbuf  = qbuf + nq;
  unsigned short* vbuf  = kbuf + nq;
  unsigned short* abf   = vbuf + nq;

  // 1) Convert inputs/weights to bf16
  convert_f32_bf16<<<(int)((nx + 255) / 256), 256, 0, stream>>>(x, xbf, (int)nx);
  convert_f32_bf16<<<(3 * EMBED * EMBED + 255) / 256, 256, 0, stream>>>(
      w_qkv, wqkvb, 3 * EMBED * EMBED);
  convert_f32_bf16<<<(EMBED * EMBED + 255) / 256, 256, 0, stream>>>(
      w_out, woutb, EMBED * EMBED);

  // 2) QKV projection: [8192 x 768] * [2304 x 768]^T -> split Q/K/V per head
  gemm_bf16_wmma<<<dim3(64, 18), 256, 0, stream>>>(
      xbf, wqkvb, b_qkv, nullptr, qbuf, kbuf, vbuf,
      (int)M, 3 * EMBED, EMBED, 0);

  // 3) Flash attention per (b,h)
  attn_wmma<<<dim3(SEQ / 128, BB * NHEAD), 256, 0, stream>>>(qbuf, kbuf, vbuf, abf);

  // 4) Output projection: [8192 x 768] * [768 x 768]^T + bias -> f32 out
  gemm_bf16_wmma<<<dim3(64, 6), 256, 0, stream>>>(
      abf, woutb, b_out, out, nullptr, nullptr, nullptr,
      (int)M, EMBED, EMBED, 1);
}
